// GNNModel_71648644432367
// MI455X (gfx1250) — matrix operand order repaired and checked by reading, not compiled, not run on hardware
//
#include <hip/hip_runtime.h>

// ---------------------------------------------------------------------------
// 2-layer GCN for MI455X (gfx1250, wave32, WMMA).
//   h1 = leakyrelu( Agg( x @ W1 ) + b1 )
//   h2 = leakyrelu( Agg( h1 @ W2 ) + b2 )
//   out = mean_rows(h2)                       [1,128] fp32
//
// GEMMs: v_wmma_f32_16x16x32_bf16 (bf16 A/B, fp32 accumulate).
//   * W is repacked ONCE into WMMA-fragment order (bf16): the per-k-step
//     B fetch becomes one contiguous 32B load instead of 16 strided b32s.
//   * Block = 256 thr = 8 waves -> 64 rows x 128 cols; each wave owns one
//     16-col stripe and 4 M-tiles, so one B fragment feeds 4 WMMAs.
// Aggregation: streaming 512B row gathers + global_atomic_add_f32 scatters
//   (~2 GB @ 23.3 TB/s dominates the runtime; GEMMs are ~11 GFLOP).
// ---------------------------------------------------------------------------

typedef __attribute__((ext_vector_type(16))) __bf16 v16bf;
typedef __attribute__((ext_vector_type(8)))  float  v8f;

#define HID 128   // hidden width (fixed by the model)
#define MT  4     // M-tiles (of 16 rows) per block

// ---------------- utility: zero a float buffer ----------------
__global__ __launch_bounds__(256) void k_zero(float* __restrict__ p, long n) {
  long i = (long)blockIdx.x * blockDim.x + threadIdx.x;
  long stride = (long)gridDim.x * blockDim.x;
  for (; i < n; i += stride) p[i] = 0.0f;
}

// ---------------- degree count (edges only; +1 self-loop added later) ------
__global__ __launch_bounds__(256) void k_deg(const int* __restrict__ dst,
                                             float* __restrict__ deg, int nE) {
  int i = blockIdx.x * blockDim.x + threadIdx.x;
  int stride = gridDim.x * blockDim.x;
  for (; i < nE; i += stride) atomicAdd(&deg[dst[i]], 1.0f);
}

// deg -> deg^{-1/2} in place (deg includes +1 self-loop, always > 0)
__global__ __launch_bounds__(256) void k_dinv(float* __restrict__ deg, int n) {
  int i = blockIdx.x * blockDim.x + threadIdx.x;
  int stride = gridDim.x * blockDim.x;
  for (; i < n; i += stride) deg[i] = rsqrtf(deg[i] + 1.0f);
}

// ---- repack W[K,128] (f32, row-major) -> Wp (bf16, WMMA B-fragment order) --
// Wp element index = ((kt*8 + ntile)*32 + lane)*16 + i, holding
//   W[kt*32 + (lane>=16)*16 + i][ntile*16 + (lane&15)]
// so each lane's 16 B values for one k-tile are 32 contiguous bytes.
__global__ __launch_bounds__(256) void k_packW(const float* __restrict__ W,
                                               __bf16* __restrict__ Wp, int K) {
  int idx = blockIdx.x * blockDim.x + threadIdx.x;
  int total = K * HID;
  if (idx >= total) return;
  int i     = idx & 15;
  int lane  = (idx >> 4) & 31;
  int ntile = (idx >> 9) & 7;
  int kt    = idx >> 12;
  int col = ntile * 16 + (lane & 15);
  int kk  = kt * 32 + (lane >> 4) * 16 + i;
  Wp[idx] = (__bf16)W[(size_t)kk * HID + col];
}

// ---------------- WMMA GEMM: C[M,128] = A[M,K] * Wp ------------------------
// Block covers 64 rows; wave w covers cols 16w..16w+15 for all 4 row-tiles.
// K must be a multiple of 32 (768 and 128 both are).
__global__ __launch_bounds__(256) void k_gemm_wmma(
    const float* __restrict__ A, const __bf16* __restrict__ Wp,
    float* __restrict__ C, int M, int K) {
  const int lane = threadIdx.x & 31;
  const int wave = threadIdx.x >> 5;          // 0..7 -> N tile
  const int row0 = blockIdx.x * (16 * MT);
  const int n0   = wave * 16;
  const int hi   = lane >> 4;                 // 0: lanes 0-15, 1: lanes 16-31

  // A-fragment base pointers per M-tile (rows clamped so EXEC stays all-1s;
  // out-of-range tiles are skipped at the store with a *uniform* branch).
  const float* arp[MT];
#pragma unroll
  for (int t = 0; t < MT; ++t) {
    int arow = row0 + 16 * t + (lane & 15);
    if (arow >= M) arow = M - 1;
    arp[t] = A + (size_t)arow * K + hi * 16;  // K offset 0 or 16
  }

  // B fragments: contiguous 32B per lane per k-tile in packed layout
  const __bf16* wpBase = Wp + (((size_t)wave * 32 + lane) << 4);

  v8f acc[MT];
#pragma unroll
  for (int t = 0; t < MT; ++t) acc[t] = (v8f)0.0f;

  const int nkt = K >> 5;                     // k-tiles of 32
  for (int kt = 0; kt < nkt; ++kt) {
    // ---- B fragment: one 32-byte vector load (shared by 4 WMMAs) ----
    v16bf b = *(const v16bf*)(wpBase + ((size_t)kt << 12));

    // ---- 4 A fragments -> 4 back-to-back WMMAs ----
#pragma unroll
    for (int t = 0; t < MT; ++t) {
      v16bf a;
      const float* ap = arp[t] + (kt << 5);
#pragma unroll
      for (int i = 0; i < 8; ++i) {
        a[i]     = (__bf16)ap[i];             // K = kt*32 + hi*16 + i
        a[8 + i] = (__bf16)ap[8 + i];         // K = kt*32 + hi*16 + 8 + i
      }
      acc[t] = __builtin_amdgcn_wmma_f32_16x16x32_bf16(
          false, a, false, b, (short)0, acc[t], false, false);
    }
  }

  // ---- store D: VGPR r -> row (r + 8*hi), col (lane&15); guards uniform ----
  const int ccol = n0 + (lane & 15);
#pragma unroll
  for (int t = 0; t < MT; ++t) {
    const int trow0 = row0 + 16 * t;
    if (trow0 + 16 <= M) {                    // full tile: branch-free stores
      float* cp = C + (size_t)(trow0 + hi * 8) * HID + ccol;
#pragma unroll
      for (int r = 0; r < 8; ++r)
        cp[(size_t)r * HID] = acc[t][r];
    } else if (trow0 < M) {                   // ragged tile (unused when M%16==0)
      const int crow0 = trow0 + hi * 8;
#pragma unroll
      for (int r = 0; r < 8; ++r)
        if (crow0 + r < M) C[(size_t)(crow0 + r) * HID + ccol] = acc[t][r];
    }
  }
}

// ---------------- edge scatter: out[d,:] += h[s,:] * dinv[s]*dinv[d] -------
// 256 threads handle 2 edges (128 columns each): streaming + f32 atomics.
__global__ __launch_bounds__(256) void k_scatter(
    const int* __restrict__ src, const int* __restrict__ dst,
    const float* __restrict__ dinv, const float* __restrict__ h,
    float* __restrict__ out, int nE) {
  int e = blockIdx.x * 2 + (threadIdx.x >> 7);
  if (e >= nE) return;
  int c = threadIdx.x & (HID - 1);
  int s = src[e], d = dst[e];
  float norm = dinv[s] * dinv[d];
  atomicAdd(&out[(size_t)d * HID + c], h[(size_t)s * HID + c] * norm);
}

// ---- h := leakyrelu(agg + h*dinv^2 + bias)  (self-loop folded in) ---------
__global__ __launch_bounds__(256) void k_post(
    const float* __restrict__ agg, float* __restrict__ h,
    const float* __restrict__ dinv, const float* __restrict__ bias, int n) {
  long total = (long)n * HID;
  long idx = (long)blockIdx.x * blockDim.x + threadIdx.x;
  long stride = (long)gridDim.x * blockDim.x;
  for (; idx < total; idx += stride) {
    int i = (int)(idx >> 7);
    int c = (int)(idx & (HID - 1));
    float di = dinv[i];
    float v = agg[idx] + h[idx] * di * di + bias[c];
    h[idx] = v > 0.0f ? v : 0.01f * v;
  }
}

// ---------------- deterministic column mean: out[c] = mean_i h[i,c] --------
__global__ __launch_bounds__(256) void k_mean(const float* __restrict__ h,
                                              float* __restrict__ out, int n) {
  __shared__ float red[256];
  int c = blockIdx.x;                         // 128 blocks, one per column
  float s = 0.0f;
  for (int i = threadIdx.x; i < n; i += 256)
    s += h[(size_t)i * HID + c];
  red[threadIdx.x] = s;
  __syncthreads();
  for (int w = 128; w > 0; w >>= 1) {
    if (threadIdx.x < w) red[threadIdx.x] += red[threadIdx.x + w];
    __syncthreads();
  }
  if (threadIdx.x == 0) out[c] = red[0] / (float)n;
}

// ---------------------------------------------------------------------------
extern "C" void kernel_launch(void* const* d_in, const int* in_sizes, int n_in,
                              void* d_out, int out_size, void* d_ws, size_t ws_size,
                              hipStream_t stream) {
  const float* x    = (const float*)d_in[0];  // [N, 768]
  const int*   eidx = (const int*)  d_in[1];  // [2, E]
  const float* W1   = (const float*)d_in[2];  // [768, 128]
  const float* b1   = (const float*)d_in[3];  // [128]
  const float* W2   = (const float*)d_in[4];  // [128, 128]
  const float* b2   = (const float*)d_in[5];  // [128]
  float*       out  = (float*)d_out;          // [1, 128]

  const int VEC = 768;
  const int nNodes = in_sizes[0] / VEC;       // 50000
  const int nE     = in_sizes[1] / 2;         // 800000
  const int* src = eidx;
  const int* dst = eidx + nE;

  // workspace: buf1 | buf2 | dinv | Wp1 | Wp2   (~51.7 MB)
  size_t nodeElems = (size_t)nNodes * HID;
  float*  buf1 = (float*)d_ws;
  float*  buf2 = buf1 + nodeElems;
  float*  dinv = buf2 + nodeElems;
  __bf16* Wp1  = (__bf16*)(dinv + nNodes);
  __bf16* Wp2  = Wp1 + (size_t)VEC * HID;

  const int mBlocks = (nNodes + 16 * MT - 1) / (16 * MT);  // 782

  // weight repack (one-shot, L2-resident afterwards)
  k_packW<<<(VEC * HID + 255) / 256, 256, 0, stream>>>(W1, Wp1, VEC);
  k_packW<<<(HID * HID + 255) / 256, 256, 0, stream>>>(W2, Wp2, HID);

  // degrees -> dinv
  k_zero<<<256, 256, 0, stream>>>(dinv, nNodes);
  k_deg <<<1024, 256, 0, stream>>>(dst, dinv, nE);
  k_dinv<<<256, 256, 0, stream>>>(dinv, nNodes);

  // ---- layer 1 ----
  k_gemm_wmma<<<mBlocks, 256, 0, stream>>>(x, Wp1, buf1, nNodes, VEC);   // h1 -> buf1
  k_zero<<<2048, 256, 0, stream>>>(buf2, (long)nodeElems);               // agg1 = 0
  k_scatter<<<(nE + 1) / 2, 256, 0, stream>>>(src, dst, dinv, buf1, buf2, nE);
  k_post<<<2048, 256, 0, stream>>>(buf2, buf1, dinv, b1, nNodes);        // act1 -> buf1

  // ---- layer 2 ----
  k_gemm_wmma<<<mBlocks, 256, 0, stream>>>(buf1, Wp2, buf2, nNodes, HID); // h2 -> buf2
  k_zero<<<2048, 256, 0, stream>>>(buf1, (long)nodeElems);                // agg2 = 0
  k_scatter<<<(nE + 1) / 2, 256, 0, stream>>>(src, dst, dinv, buf2, buf1, nE);
  k_post<<<2048, 256, 0, stream>>>(buf1, buf2, dinv, b2, nNodes);         // act2 -> buf2

  // ---- mean over nodes ----
  k_mean<<<HID, 256, 0, stream>>>(buf2, out, nNodes);
}
